// Conv_DCFD_tf_59940563583899
// MI455X (gfx1250) — matrix operand order repaired and checked by reading, not compiled, not run on hardware
//
#include <hip/hip_runtime.h>

typedef __bf16 bf16;
typedef __attribute__((ext_vector_type(16))) bf16  v16bf;
typedef __attribute__((ext_vector_type(8)))  bf16  v8bf;
typedef __attribute__((ext_vector_type(8)))  float v8f;

#define NB   16
#define HH   64
#define WW   64
#define CIN  128
#define CMID 64
#define BSZ  36      // M*BASES_LEN, padded to 48 for 3 N-tiles
#define MMB  6
#define OO   256
#define K1   (9*CIN)   // 1152
#define K2   (9*CMID)  // 576
#define K3   (CIN*MMB) // 768

__device__ inline bf16 f2bf(float f) {
  union { float f; unsigned u; } in; in.f = f;
  unsigned r = (in.u + 0x7FFFu + ((in.u >> 16) & 1u)) >> 16;
  union { unsigned short s; bf16 b; } out; out.s = (unsigned short)r;
  return out.b;
}
__device__ inline float bf2f(bf16 b) {
  union { bf16 b; unsigned short s; } in; in.b = b;
  union { unsigned u; float f; } out; out.u = ((unsigned)in.s) << 16;
  return out.f;
}

__device__ inline v8f wmma_bf16(v16bf a, v16bf b, v8f c) {
  return __builtin_amdgcn_wmma_f32_16x16x32_bf16(false, a, false, b, (short)0, c,
                                                 false, false);
}

// ---------------------------------------------------------------- prep ----
// Transpose+convert weights to bf16, K-contiguous per output column (B layout).
__global__ void k_prep(const float* __restrict__ w1, const float* __restrict__ w2,
                       const float* __restrict__ coef,
                       bf16* __restrict__ w1t, bf16* __restrict__ w2t,
                       bf16* __restrict__ coeft) {
  int i = blockIdx.x * blockDim.x + threadIdx.x;
  if (i < CMID * K1) { int n = i / K1, k = i % K1; w1t[i] = f2bf(w1[k * CMID + n]); }
  if (i < 48 * K2)   { int n = i / K2, k = i % K2;
                       w2t[i] = f2bf(n < BSZ ? w2[k * BSZ + n] : 0.f); }
  if (i < OO * K3)   { int o = i / K3, k = i % K3; coeft[i] = f2bf(coef[k * OO + o]); }
}

// --------------------------------------------------------------- conv1 ----
// One workgroup (4 waves) per (n,h) row: 64 pixels, 64 outputs, K=1152.
__global__ __launch_bounds__(128) void k_conv1(
    const float* __restrict__ x, const bf16* __restrict__ w1t,
    const float* __restrict__ b1, const float* __restrict__ g1,
    const float* __restrict__ bt1, bf16* __restrict__ hfeat) {
  __shared__ bf16 xs[3 * 66 * CIN];             // 49.5 KB halo tile, bf16
  const int tid  = threadIdx.x;
  const int nimg = blockIdx.x / HH;
  const int h    = blockIdx.x % HH;

  for (int idx = tid; idx < 3 * 66 * CIN; idx += 128) {
    int r = idx / (66 * CIN), rem = idx % (66 * CIN);
    int col = rem / CIN, c = rem % CIN;
    int hh = h + r - 1, ww = col - 1;
    float v = 0.f;
    if (hh >= 0 && hh < HH && ww >= 0 && ww < WW)
      v = x[((nimg * HH + hh) * WW + ww) * CIN + c];
    xs[idx] = f2bf(v);
  }
  __syncthreads();

  const int wave = tid >> 5, lane = tid & 31;
  const int lhalf = lane >> 4, l16 = lane & 15;
  const int p0 = wave * 16;

  v8f acc[4];
  #pragma unroll
  for (int j = 0; j < 4; ++j)
    #pragma unroll
    for (int r = 0; r < 8; ++r) acc[j][r] = 0.f;

  for (int t = 0; t < K1 / 32; ++t) {
    const int kk = t * 32;
    const int ki = kk / (3 * CIN), kj = (kk % (3 * CIN)) / CIN, c0 = kk % CIN;
    const int abase = (ki * 66 + p0 + l16 + kj) * CIN + c0 + lhalf * 16;
    union { v16bf v; v8bf h[2]; } a;
    a.h[0] = *(const v8bf*)&xs[abase];
    a.h[1] = *(const v8bf*)&xs[abase + 8];
    #pragma unroll
    for (int nt = 0; nt < 4; ++nt) {
      v16bf b = *(const v16bf*)&w1t[(nt * 16 + l16) * K1 + kk + lhalf * 16];
      acc[nt] = wmma_bf16(a.v, b, acc[nt]);
    }
  }
  const float rs = rsqrtf(1.0f + 1e-3f);
  const float g = g1[h] * rs, bt = bt1[h];
  #pragma unroll
  for (int nt = 0; nt < 4; ++nt) {
    const int n = nt * 16 + l16;
    const float bias = b1[n];
    #pragma unroll
    for (int r = 0; r < 8; ++r) {
      const int p = p0 + r + lhalf * 8;
      float v = g * (acc[nt][r] + bias) + bt;
      v = v > 0.f ? v : 0.f;
      hfeat[(blockIdx.x * WW + p) * CMID + n] = f2bf(v);
    }
  }
}

// --------------------------------------------------------------- conv2 ----
// conv2 + BN + ReLU, then fused bfeat[·,6,6] @ FB[6,9] -> dyn [·,54] bf16.
__global__ __launch_bounds__(128) void k_conv2(
    const bf16* __restrict__ hfeat, const bf16* __restrict__ w2t,
    const float* __restrict__ b2, const float* __restrict__ g2,
    const float* __restrict__ bt2, const float* __restrict__ fb,
    bf16* __restrict__ dyn) {
  __shared__ bf16  hs[3 * 66 * CMID];   // 24.75 KB
  __shared__ float bfs[4][16 * 48];     // 12 KB (per-wave bfeat tile)
  __shared__ float fbs[54];
  const int tid  = threadIdx.x;
  const int nimg = blockIdx.x / HH;
  const int h    = blockIdx.x % HH;
  if (tid < 54) fbs[tid] = fb[tid];

  for (int idx = tid; idx < 3 * 66 * CMID; idx += 128) {
    int r = idx / (66 * CMID), rem = idx % (66 * CMID);
    int col = rem / CMID, c = rem % CMID;
    int hh = h + r - 1, ww = col - 1;
    bf16 v = f2bf(0.f);
    if (hh >= 0 && hh < HH && ww >= 0 && ww < WW)
      v = hfeat[((nimg * HH + hh) * WW + ww) * CMID + c];
    hs[idx] = v;
  }
  __syncthreads();

  const int wave = tid >> 5, lane = tid & 31;
  const int lhalf = lane >> 4, l16 = lane & 15;
  const int p0 = wave * 16;

  v8f acc[3];
  #pragma unroll
  for (int j = 0; j < 3; ++j)
    #pragma unroll
    for (int r = 0; r < 8; ++r) acc[j][r] = 0.f;

  for (int t = 0; t < K2 / 32; ++t) {
    const int kk = t * 32;
    const int ki = kk / (3 * CMID), kj = (kk % (3 * CMID)) / CMID, c0 = kk % CMID;
    const int abase = (ki * 66 + p0 + l16 + kj) * CMID + c0 + lhalf * 16;
    union { v16bf v; v8bf h[2]; } a;
    a.h[0] = *(const v8bf*)&hs[abase];
    a.h[1] = *(const v8bf*)&hs[abase + 8];
    #pragma unroll
    for (int nt = 0; nt < 3; ++nt) {
      v16bf b = *(const v16bf*)&w2t[(nt * 16 + l16) * K2 + kk + lhalf * 16];
      acc[nt] = wmma_bf16(a.v, b, acc[nt]);
    }
  }
  const float rs = rsqrtf(1.0f + 1e-3f);
  const float g = g2[h] * rs, bt = bt2[h];
  #pragma unroll
  for (int nt = 0; nt < 3; ++nt) {
    const int n = nt * 16 + l16;
    const float bias = (n < BSZ) ? b2[n] : 0.f;
    #pragma unroll
    for (int r = 0; r < 8; ++r) {
      const int p = r + lhalf * 8;              // local pixel 0..15
      float v = g * (acc[nt][r] + bias) + bt;
      v = v > 0.f ? v : 0.f;
      if (n >= BSZ) v = 0.f;
      bfs[wave][p * 48 + n] = v;
    }
  }
  __syncthreads();
  // dyn[p, m, l] = sum_t bfeat[p, m*6+t] * FB[t, l]
  for (int idx = lane; idx < 16 * 54; idx += 32) {
    int p = idx / 54, rem = idx % 54;
    int m = rem / 9, l = rem % 9;
    float v = 0.f;
    #pragma unroll
    for (int t2 = 0; t2 < 6; ++t2)
      v += bfs[wave][p * 48 + m * 6 + t2] * fbs[t2 * 9 + l];
    dyn[(blockIdx.x * WW + p0 + p) * 54 + rem] = f2bf(v);
  }
}

// --------------------------------------------------------------- final ----
// Per 16-pixel tile: build bases_out [16,768] bf16 in LDS (never hits HBM),
// then GEMM against coefT [256,768] bf16 with WMMA. 8 waves × 2 N-tiles.
__global__ __launch_bounds__(256) void k_final(
    const float* __restrict__ x, const bf16* __restrict__ dyn,
    const bf16* __restrict__ coeft, const float* __restrict__ bias,
    float* __restrict__ out) {
  __shared__ float xs[3 * 18 * CIN];   // 27 KB fp32 halo
  __shared__ float dy[16 * 54];        // 3.4 KB
  __shared__ bf16  bo[16 * K3];        // 24 KB bases_out tile
  const int tid  = threadIdx.x;
  const int P0   = blockIdx.x * 16;
  const int nimg = P0 / (HH * WW);
  const int h    = (P0 / WW) % HH;
  const int w0   = P0 % WW;

  for (int idx = tid; idx < 3 * 18 * CIN; idx += 256) {
    int r = idx / (18 * CIN), rem = idx % (18 * CIN);
    int col = rem / CIN, c = rem % CIN;
    int hh = h + r - 1, ww = w0 + col - 1;
    float v = 0.f;
    if (hh >= 0 && hh < HH && ww >= 0 && ww < WW)
      v = x[((nimg * HH + hh) * WW + ww) * CIN + c];
    xs[idx] = v;
  }
  for (int idx = tid; idx < 16 * 54; idx += 256)
    dy[idx] = bf2f(dyn[P0 * 54 + idx]);
  __syncthreads();

  // bases_out[p, c*6+m] = sum_l patch(x)[p,c,l] * dyn[p,m,l]
  for (int idx = tid; idx < 16 * K3; idx += 256) {
    int p = idx / K3, cm = idx % K3;
    int c = cm / MMB, m = cm % MMB;
    float v = 0.f;
    #pragma unroll
    for (int l = 0; l < 9; ++l) {
      int ki = l / 3, kj = l % 3;
      v += xs[(ki * 18 + p + kj) * CIN + c] * dy[p * 54 + m * 9 + l];
    }
    bo[idx] = f2bf(v);
  }
  __syncthreads();

  const int wave = tid >> 5, lane = tid & 31;
  const int lhalf = lane >> 4, l16 = lane & 15;

  v8f acc[2];
  #pragma unroll
  for (int j = 0; j < 2; ++j)
    #pragma unroll
    for (int r = 0; r < 8; ++r) acc[j][r] = 0.f;

  for (int t = 0; t < K3 / 32; ++t) {
    const int kk = t * 32;
    const int abase = l16 * K3 + kk + lhalf * 16;
    union { v16bf v; v8bf h[2]; } a;
    a.h[0] = *(const v8bf*)&bo[abase];
    a.h[1] = *(const v8bf*)&bo[abase + 8];
    #pragma unroll
    for (int j = 0; j < 2; ++j) {
      const int n = (wave * 2 + j) * 16 + l16;
      v16bf b = *(const v16bf*)&coeft[n * K3 + kk + lhalf * 16];
      acc[j] = wmma_bf16(a.v, b, acc[j]);
    }
  }
  #pragma unroll
  for (int j = 0; j < 2; ++j) {
    const int o = (wave * 2 + j) * 16 + l16;
    const float bs = bias[o];
    #pragma unroll
    for (int r = 0; r < 8; ++r) {
      const int p = r + lhalf * 8;
      out[(P0 + p) * OO + o] = acc[j][r] + bs;
    }
  }
}

// -------------------------------------------------------------- launch ----
extern "C" void kernel_launch(void* const* d_in, const int* in_sizes, int n_in,
                              void* d_out, int out_size, void* d_ws, size_t ws_size,
                              hipStream_t stream) {
  const float* x    = (const float*)d_in[0];
  const float* w1   = (const float*)d_in[1];
  const float* b1   = (const float*)d_in[2];
  const float* g1   = (const float*)d_in[3];
  const float* bt1  = (const float*)d_in[4];
  const float* w2   = (const float*)d_in[5];
  const float* b2   = (const float*)d_in[6];
  const float* g2   = (const float*)d_in[7];
  const float* bt2  = (const float*)d_in[8];
  const float* fb   = (const float*)d_in[9];
  const float* coef = (const float*)d_in[10];
  const float* bias = (const float*)d_in[11];
  float* out = (float*)d_out;

  char* ws = (char*)d_ws;                     // all offsets 256B aligned
  bf16* w1t   = (bf16*)(ws + 0);              // 64*1152*2   = 147456
  bf16* w2t   = (bf16*)(ws + 147456);         // 48*576*2    =  55296
  bf16* coeft = (bf16*)(ws + 202752);         // 256*768*2   = 393216
  bf16* hfeat = (bf16*)(ws + 595968);         // 16*64*64*64*2 = 8388608
  bf16* dyn   = (bf16*)(ws + 8984576);        // 16*64*64*54*2 = 7077888

  k_prep <<<(OO * K3 + 255) / 256, 256, 0, stream>>>(w1, w2, coef, w1t, w2t, coeft);
  k_conv1<<<NB * HH, 128, 0, stream>>>(x, w1t, b1, g1, bt1, hfeat);
  k_conv2<<<NB * HH, 128, 0, stream>>>(hfeat, w2t, b2, g2, bt2, fb, dyn);
  k_final<<<NB * HH * WW / 16, 256, 0, stream>>>(x, dyn, coeft, bias, out);
}
